// DABConv_60473139528289
// MI455X (gfx1250) — matrix operand order repaired and checked by reading, not compiled, not run on hardware
//
#include <hip/hip_runtime.h>

// ---------------------------------------------------------------------------
// DABConv for MI455X (gfx1250, wave32, WMMA + async LDS copies).
// Pipeline: pack -> conv_std(bf16 wmma) -> conv_offmod(f32 wmma) ->
//           deform(gather + bf16 wmma) -> fuse(bf16 wmma).
// Weight panels are staged into LDS with GLOBAL_LOAD_ASYNC_TO_LDS_B128
// (ASYNCcnt / s_wait_asynccnt), activation tiles go through VGPRs because
// they need pad-masking / f32->bf16 packing.
// ---------------------------------------------------------------------------

typedef __attribute__((ext_vector_type(16))) __bf16 v16bf;
typedef __attribute__((ext_vector_type(8)))  float  v8f;
typedef __attribute__((ext_vector_type(2)))  float  v2f;

#define IMG_H   64
#define IMG_W   64
#define HW      4096
#define CIN     128
#define COUT    128
#define NTILES  2048         // B*HW/16 16-pixel N tiles (one row segment each)

// ---- workspace layout (bytes) ----
#define OFF_YSTD 0u                                   // f32 [B][128][HW]
#define OFF_YOM  (OFF_YSTD + 8u*128u*4096u*4u)        // f32 [B][32][HW] (0-17 off,18-30 mod)
#define OFF_XDEF (OFF_YOM  + 8u*32u*4096u*4u)         // f32 [B][128][HW]
#define OFF_WSTD (OFF_XDEF + 8u*128u*4096u*4u)        // bf16 [9][128][128]
#define OFF_WDEF (OFF_WSTD + 9u*128u*128u*2u)         // bf16 [9][128][128]
#define OFF_WOM  (OFF_WDEF + 9u*128u*128u*2u)         // f32  [9][32][128]
#define OFF_WFUS (OFF_WOM  + 9u*32u*128u*4u)          // bf16 [128][256]

#if defined(__has_builtin)
#  if __has_builtin(__builtin_amdgcn_wmma_f32_16x16x4_f32)
#    define HAVE_WMMA_F32 1
#  else
#    define HAVE_WMMA_F32 0
#  endif
#else
#  define HAVE_WMMA_F32 0
#endif

__device__ __forceinline__ unsigned short f2bf(float f) {
  unsigned u = __float_as_uint(f);
  return (unsigned short)((u + 0x7FFFu + ((u >> 16) & 1u)) >> 16);  // RNE-ish
}
__device__ __forceinline__ unsigned pack2bf(float lo, float hi) {
  return (unsigned)f2bf(lo) | ((unsigned)f2bf(hi) << 16);
}
__device__ __forceinline__ float sigmoidf(float x) {
  return 1.0f / (1.0f + __expf(-x));
}

// ---- CDNA5 async copy: global -> LDS, 16B per lane, tracked by ASYNCcnt ----
__device__ __forceinline__ void async_b128_to_lds(unsigned lds_byte, const void* g) {
  asm volatile("global_load_async_to_lds_b128 %0, %1, off"
               :: "v"(lds_byte), "v"((unsigned long long)(__SIZE_TYPE__)g)
               : "memory");
}
__device__ __forceinline__ void wait_async0() {
  asm volatile("s_wait_asynccnt 0" ::: "memory");
}
__device__ __forceinline__ unsigned lds_addr_of(const void* p) {
  return (unsigned)(__SIZE_TYPE__)p;        // low 32 bits of LDS aperture addr
}

// ---------------------------------------------------------------------------
// Kernel 0: repack weights into k-major GEMM layouts.
// ---------------------------------------------------------------------------
__global__ void k_pack(const float* __restrict__ Wstd, const float* __restrict__ Woff,
                       const float* __restrict__ Wmod, const float* __restrict__ Wdef,
                       const float* __restrict__ Wfus,
                       unsigned short* __restrict__ pWstd, float* __restrict__ pWom,
                       unsigned short* __restrict__ pWdef, unsigned short* __restrict__ pWfus) {
  const int N1 = 9 * 128 * 128;
  const int N2 = 9 * 32 * 128;
  const int N3 = 9 * 128 * 128;
  const int N4 = 128 * 256;
  int i = blockIdx.x * blockDim.x + threadIdx.x;
  if (i < N1) {                      // W_std (O,C,3,3) -> [k][o][c] bf16
    int k = i / (128 * 128), r = i % (128 * 128), o = r >> 7, c = r & 127;
    pWstd[i] = f2bf(Wstd[(o * 128 + c) * 9 + k]);
  } else if (i < N1 + N2) {          // W_off(18)+W_mod(13)+pad -> [k][32][c] f32
    int j = i - N1;
    int k = j / (32 * 128), r = j % (32 * 128), oc = r >> 7, c = r & 127;
    float v = 0.f;
    if (oc < 18)      v = Woff[(oc * 128 + c) * 9 + k];
    else if (oc < 31) v = Wmod[((oc - 18) * 128 + c) * 9 + k];
    pWom[j] = v;
  } else if (i < N1 + N2 + N3) {     // W_def -> [k][o][c] bf16
    int j = i - N1 - N2;
    int k = j / (128 * 128), r = j % (128 * 128), o = r >> 7, c = r & 127;
    pWdef[j] = f2bf(Wdef[(o * 128 + c) * 9 + k]);
  } else if (i < N1 + N2 + N3 + N4) { // W_fus (128,256) -> bf16
    int j = i - N1 - N2 - N3;
    pWfus[j] = f2bf(Wfus[j]);
  }
}

// ---------------------------------------------------------------------------
// bf16 WMMA fragment step (A 16x32 / B 32x16 per ISA tables; LDS holds packed
// bf16 pairs with padded row stride 65 dwords => conflict-free).
// ---------------------------------------------------------------------------
__device__ __forceinline__ v8f wmma_bf16_step(const unsigned* As, const unsigned* Bs,
                                              int mrow, int lane, int c0, v8f acc) {
  const bool hi = lane >= 16;
  const int  cu = c0 >> 1;
  union { v16bf v; unsigned u[8]; } A, B;
  const int aoff = mrow * 65 + cu + (hi ? 4 : 0);
#pragma unroll
  for (int j = 0; j < 4; ++j) {       // K = 2j(+8 hi) and K = 16+2j(+8 hi)
    A.u[j]     = As[aoff + j];
    A.u[j + 4] = As[aoff + 8 + j];
  }
  const int boff = (lane & 15) * 65 + cu + (hi ? 4 : 0);   // K = 2j(+8 hi) and K = 16+2j(+8 hi)
#pragma unroll
  for (int j = 0; j < 4; ++j) {
    B.u[j]     = Bs[boff + j];
    B.u[j + 4] = Bs[boff + 8 + j];
  }
  return __builtin_amdgcn_wmma_f32_16x16x32_bf16(false, A.v, false, B.v,
                                                 (short)0, acc, false, false);
}

// Async-stage a [128][64]-dword weight panel into As (row stride 65 dwords).
// 8 x b128 per thread, 256 threads. 4 | 64 so no b128 crosses a padded row.
__device__ __forceinline__ void stage_panel_async(unsigned* As, const unsigned* src,
                                                  int t) {
  const unsigned base = lds_addr_of(As);
#pragma unroll
  for (int j = 0; j < 8; ++j) {
    const int i = (t + (j << 8)) << 2;          // dword idx, 16B aligned
    const int o = i >> 6, cc = i & 63;
    async_b128_to_lds(base + (unsigned)((o * 65 + cc) << 2), src + i);
  }
}

// ---------------------------------------------------------------------------
// Kernel 1: standard 3x3 conv (128->128) as implicit GEMM, bf16 WMMA.
// Block = 256 (8 waves = 8 M-tiles of 16 channels); grid = 2048 N-tiles.
// ---------------------------------------------------------------------------
__global__ __launch_bounds__(256) void k_conv_std(const float* __restrict__ x,
                                                  const unsigned* __restrict__ Wp,  // [9][128][64] u32
                                                  const float* __restrict__ bias,
                                                  float* __restrict__ y) {
  __shared__ unsigned As[128 * 65];
  __shared__ unsigned Bs[16 * 65];
  const int tile = blockIdx.x;
  const int b  = tile >> 8;
  const int pb = (tile & 255) << 4;
  const int h  = pb >> 6, w0 = pb & 63;
  const int t = threadIdx.x, lane = t & 31, mt = t >> 5;
  const int mrow = (mt << 4) + (lane & 15);
  const bool hi = lane >= 16;
  v8f acc = {};

  for (int kk = 0; kk < 9; ++kk) {
    const int kh = kk / 3, kw = kk % 3;
    if (kk < 8) __builtin_prefetch(Wp + (kk + 1) * (128 * 64), 0, 0);
    stage_panel_async(As, Wp + kk * (128 * 64), t);   // weights -> LDS (async)
    {                                                 // B tile: 16 pix x 128 ch
      const int n = t >> 4, cb = (t & 15) << 3;
      const int yy = h + kh - 1;
      const int xc = w0 + n + kw - 1;
      const bool ok = (yy >= 0) & (yy < IMG_H) & (xc >= 0) & (xc < IMG_W);
#pragma unroll
      for (int cj = 0; cj < 8; cj += 2) {
        const int c = cb + cj;
        float f0 = 0.f, f1 = 0.f;
        if (ok) {
          const float* px = x + ((size_t)(b * CIN + c)) * HW + yy * IMG_W + xc;
          f0 = px[0]; f1 = px[HW];
        }
        Bs[n * 65 + (c >> 1)] = pack2bf(f0, f1);
      }
    }
    wait_async0();
    __syncthreads();
#pragma unroll
    for (int c0 = 0; c0 < 128; c0 += 32)
      acc = wmma_bf16_step(As, Bs, mrow, lane, c0, acc);
    __syncthreads();
  }
  const int col = w0 + (lane & 15);
#pragma unroll
  for (int v = 0; v < 8; ++v) {                   // f32 C/D: M = v (+8 hi), N = lane&15
    const int ch = (mt << 4) + v + (hi ? 8 : 0);
    y[((size_t)(b * COUT + ch)) * HW + h * IMG_W + col] = acc[v] + bias[ch];
  }
}

// ---------------------------------------------------------------------------
// Kernel 2: offset(18)+modulator(13) conv in FULL FP32 via V_WMMA_F32_16X16X4_F32
// (offsets feed sampling positions -> keep precision). Block = 64 (2 waves).
// ---------------------------------------------------------------------------
__global__ __launch_bounds__(64) void k_conv_om(const float* __restrict__ x,
                                                const float* __restrict__ Wom,  // [9][32][128]
                                                const float* __restrict__ b_off,
                                                const float* __restrict__ b_mod,
                                                float* __restrict__ yom) {
  __shared__ float As[32 * 129];
  __shared__ float Bs[16 * 129];
  const int tile = blockIdx.x;
  const int b  = tile >> 8;
  const int pb = (tile & 255) << 4;
  const int h  = pb >> 6, w0 = pb & 63;
  const int t = threadIdx.x, lane = t & 31, mt = t >> 5;   // mt 0..1
  const int mrow = (mt << 4) + (lane & 15);
  const bool hi = lane >= 16;
  v8f acc = {};

  for (int kk = 0; kk < 9; ++kk) {
    const int kh = kk / 3, kw = kk % 3;
    {                                             // stage A async: 4096 f32
      const float* src = Wom + kk * (32 * 128);
      const unsigned base = lds_addr_of(As);
#pragma unroll
      for (int j = 0; j < 16; ++j) {
        const int i = (t + (j << 6)) << 2;        // 4 | 128: stays in one row
        const int o = i >> 7, cc = i & 127;
        async_b128_to_lds(base + (unsigned)((o * 129 + cc) << 2), src + i);
      }
    }
    {                                             // stage B: 2048 f32
      const int yy = h + kh - 1;
#pragma unroll
      for (int j = 0; j < 32; ++j) {
        int i = (t << 5) + j;
        int n = i >> 7, c = i & 127;
        int xc = w0 + n + kw - 1;
        bool ok = (yy >= 0) & (yy < IMG_H) & (xc >= 0) & (xc < IMG_W);
        Bs[n * 129 + c] = ok ? x[((size_t)(b * CIN + c)) * HW + yy * IMG_W + xc] : 0.f;
      }
    }
    wait_async0();
    __syncthreads();
#pragma unroll
    for (int c0 = 0; c0 < 128; c0 += 4) {
#if HAVE_WMMA_F32
      v2f a, bb;
      const int base = c0 + (hi ? 2 : 0);         // f32 A 16x4 / B 4x16 layout
      a.x  = As[mrow * 129 + base];       a.y  = As[mrow * 129 + base + 1];
      bb.x = Bs[(lane & 15) * 129 + base]; bb.y = Bs[(lane & 15) * 129 + base + 1];
      acc = __builtin_amdgcn_wmma_f32_16x16x4_f32(false, a, false, bb,
                                                  (short)0, acc, false, false);
#else
#pragma unroll
      for (int v = 0; v < 8; ++v) {
        const int m = (mt << 4) + v + (hi ? 8 : 0), n = lane & 15;
#pragma unroll
        for (int kq = 0; kq < 4; ++kq)
          acc[v] += As[m * 129 + c0 + kq] * Bs[n * 129 + c0 + kq];
      }
#endif
    }
    __syncthreads();
  }
  const int pix = h * IMG_W + w0 + (lane & 15);
#pragma unroll
  for (int v = 0; v < 8; ++v) {
    const int ch = (mt << 4) + v + (hi ? 8 : 0);
    float bsv = (ch < 18) ? b_off[ch] : ((ch < 31) ? b_mod[ch - 18] : 0.f);
    yom[((size_t)(b * 32 + ch)) * HW + pix] = acc[v] + bsv;
  }
}

// ---------------------------------------------------------------------------
// Kernel 3: deformable sampling + einsum (M=128, K=C*9) with bf16 WMMA.
// ---------------------------------------------------------------------------
__global__ __launch_bounds__(256) void k_deform(const float* __restrict__ x,
                                                const float* __restrict__ yom,
                                                const unsigned* __restrict__ Wdp,  // [9][128][64] u32
                                                const float* __restrict__ b_def,
                                                float* __restrict__ xdef) {
  __shared__ unsigned As[128 * 65];
  __shared__ unsigned Bs[16 * 65];
  __shared__ float s_y0[144], s_x0[144], s_wy[144], s_wx[144], s_m[144];
  const int tile = blockIdx.x;
  const int b  = tile >> 8;
  const int pb = (tile & 255) << 4;
  const int h  = pb >> 6, w0 = pb & 63;
  const int t = threadIdx.x, lane = t & 31, mt = t >> 5;
  const int mrow = (mt << 4) + (lane & 15);
  const bool hi = lane >= 16;

  if (t < 144) {                                  // per (pixel, tap) sampling state
    const int n = t / 9, k = t % 9;
    const int pix = h * IMG_W + w0 + n;
    const float* yo = yom + ((size_t)b * 32) * HW + pix;
    const float dy = yo[(2 * k) * HW];
    const float dx = yo[(2 * k + 1) * HW];
    float m = sigmoidf(yo[(18 + k) * HW]);
    const int ci = (k == 0) ? 0 : (k == 2) ? 1 : (k == 6) ? 2 : (k == 8) ? 3 : -1;
    m *= (ci >= 0) ? sigmoidf(yo[(27 + ci) * HW]) : 0.5f;
    const float py = (float)(h + k / 3 - 1) + dy;
    const float px = (float)(w0 + n + k % 3 - 1) + dx;
    const float fy = floorf(py), fx = floorf(px);
    s_y0[t] = fy; s_x0[t] = fx; s_wy[t] = py - fy; s_wx[t] = px - fx; s_m[t] = m;
  }
  __syncthreads();

  v8f acc = {};
  for (int k = 0; k < 9; ++k) {
    if (k < 8) __builtin_prefetch(Wdp + (k + 1) * (128 * 64), 0, 0);
    stage_panel_async(As, Wdp + k * (128 * 64), t);   // W_def panel -> LDS (async)
    {                                             // bilinear gather -> bf16 B tile
      const int n = t >> 4, cb = (t & 15) << 3;
      const int s = n * 9 + k;
      const float wy = s_wy[s], wx = s_wx[s], m = s_m[s];
      const int iy0 = (int)s_y0[s], ix0 = (int)s_x0[s];
      const int iy1 = iy0 + 1, ix1 = ix0 + 1;
      const bool vy0 = (iy0 >= 0) & (iy0 < IMG_H), vy1 = (iy1 >= 0) & (iy1 < IMG_H);
      const bool vx0 = (ix0 >= 0) & (ix0 < IMG_W), vx1 = (ix1 >= 0) & (ix1 < IMG_W);
      const int cy0 = min(max(iy0, 0), IMG_H - 1), cy1 = min(max(iy1, 0), IMG_H - 1);
      const int cx0 = min(max(ix0, 0), IMG_W - 1), cx1 = min(max(ix1, 0), IMG_W - 1);
      const int i00 = cy0 * IMG_W + cx0, i01 = cy0 * IMG_W + cx1;
      const int i10 = cy1 * IMG_W + cx0, i11 = cy1 * IMG_W + cx1;
      const float f00 = (vy0 & vx0) ? (1.f - wy) * (1.f - wx) * m : 0.f;
      const float f01 = (vy0 & vx1) ? (1.f - wy) * wx * m : 0.f;
      const float f10 = (vy1 & vx0) ? wy * (1.f - wx) * m : 0.f;
      const float f11 = (vy1 & vx1) ? wy * wx * m : 0.f;
#pragma unroll
      for (int cj = 0; cj < 8; cj += 2) {
        float r[2];
#pragma unroll
        for (int q = 0; q < 2; ++q) {
          const float* xc = x + ((size_t)(b * CIN + cb + cj + q)) * HW;
          r[q] = xc[i00] * f00 + xc[i01] * f01 + xc[i10] * f10 + xc[i11] * f11;
        }
        Bs[n * 65 + ((cb + cj) >> 1)] = pack2bf(r[0], r[1]);
      }
    }
    wait_async0();
    __syncthreads();
#pragma unroll
    for (int c0 = 0; c0 < 128; c0 += 32)
      acc = wmma_bf16_step(As, Bs, mrow, lane, c0, acc);
    __syncthreads();
  }
  const int col = w0 + (lane & 15);
#pragma unroll
  for (int v = 0; v < 8; ++v) {
    const int ch = (mt << 4) + v + (hi ? 8 : 0);
    xdef[((size_t)(b * COUT + ch)) * HW + h * IMG_W + col] = acc[v] + b_def[ch];
  }
}

// ---------------------------------------------------------------------------
// Kernel 4: 1x1 fusion conv on concat(y_std, x_def): M=128, K=256, bf16 WMMA.
// A fragments read straight from L2-resident packed W_fus.
// ---------------------------------------------------------------------------
__global__ __launch_bounds__(256) void k_fuse(const float* __restrict__ ystd,
                                              const float* __restrict__ xdef,
                                              const unsigned* __restrict__ Wfu,  // [128][128] u32
                                              const float* __restrict__ b_fus,
                                              float* __restrict__ out) {
  __shared__ unsigned Bs[16 * 129];
  const int tile = blockIdx.x;
  const int b  = tile >> 8;
  const int pb = (tile & 255) << 4;
  const int h  = pb >> 6, w0 = pb & 63;
  const int t = threadIdx.x, lane = t & 31, mt = t >> 5;
  const int mrow = (mt << 4) + (lane & 15);
  const bool hi = lane >= 16;
  {                                               // stage B: 16 pix x 256 ch bf16
    const int n = t >> 4, cb = (t & 15) << 4;
    const int pix = h * IMG_W + w0 + n;
#pragma unroll
    for (int cj = 0; cj < 16; cj += 2) {
      const int c = cb + cj;
      float f0 = (c < 128) ? ystd[((size_t)(b * 128 + c)) * HW + pix]
                           : xdef[((size_t)(b * 128 + c - 128)) * HW + pix];
      float f1 = (c + 1 < 128) ? ystd[((size_t)(b * 128 + c + 1)) * HW + pix]
                               : xdef[((size_t)(b * 128 + c - 127)) * HW + pix];
      Bs[n * 129 + (c >> 1)] = pack2bf(f0, f1);
    }
  }
  __syncthreads();
  v8f acc = {};
  const unsigned* arow = Wfu + mrow * 128;
#pragma unroll
  for (int c0 = 0; c0 < 256; c0 += 32) {
    const int cu = c0 >> 1;
    union { v16bf v; unsigned u[8]; } A, B;
#pragma unroll
    for (int j = 0; j < 4; ++j) {
      A.u[j]     = arow[cu + j + (hi ? 4 : 0)];
      A.u[j + 4] = arow[cu + 8 + j + (hi ? 4 : 0)];
    }
    const int boff = (lane & 15) * 129 + cu + (hi ? 4 : 0);
#pragma unroll
    for (int j = 0; j < 4; ++j) {
      B.u[j]     = Bs[boff + j];
      B.u[j + 4] = Bs[boff + 8 + j];
    }
    acc = __builtin_amdgcn_wmma_f32_16x16x32_bf16(false, A.v, false, B.v,
                                                  (short)0, acc, false, false);
  }
  const int col = w0 + (lane & 15);
#pragma unroll
  for (int v = 0; v < 8; ++v) {
    const int ch = (mt << 4) + v + (hi ? 8 : 0);
    out[((size_t)(b * COUT + ch)) * HW + h * IMG_W + col] = acc[v] + b_fus[ch];
  }
}

// ---------------------------------------------------------------------------
extern "C" void kernel_launch(void* const* d_in, const int* in_sizes, int n_in,
                              void* d_out, int out_size, void* d_ws, size_t ws_size,
                              hipStream_t stream) {
  (void)in_sizes; (void)n_in; (void)out_size; (void)ws_size;
  const float* x     = (const float*)d_in[0];
  const float* W_std = (const float*)d_in[1];
  const float* b_std = (const float*)d_in[2];
  const float* W_off = (const float*)d_in[3];
  const float* b_off = (const float*)d_in[4];
  // d_in[5]=W_corner, d_in[6]=b_corner: unused by the reference forward.
  const float* W_mod = (const float*)d_in[7];
  const float* b_mod = (const float*)d_in[8];
  const float* W_def = (const float*)d_in[9];
  const float* b_def = (const float*)d_in[10];
  const float* W_fus = (const float*)d_in[11];
  const float* b_fus = (const float*)d_in[12];

  char* ws = (char*)d_ws;
  float*          y_std = (float*)(ws + OFF_YSTD);
  float*          y_om  = (float*)(ws + OFF_YOM);
  float*          x_def = (float*)(ws + OFF_XDEF);
  unsigned short* pWstd = (unsigned short*)(ws + OFF_WSTD);
  unsigned short* pWdef = (unsigned short*)(ws + OFF_WDEF);
  float*          pWom  = (float*)(ws + OFF_WOM);
  unsigned short* pWfus = (unsigned short*)(ws + OFF_WFUS);

  const int npack = 9 * 128 * 128 + 9 * 32 * 128 + 9 * 128 * 128 + 128 * 256;
  k_pack<<<(npack + 255) / 256, 256, 0, stream>>>(W_std, W_off, W_mod, W_def, W_fus,
                                                  pWstd, pWom, pWdef, pWfus);
  k_conv_std<<<NTILES, 256, 0, stream>>>(x, (const unsigned*)pWstd, b_std, y_std);
  k_conv_om<<<NTILES, 64, 0, stream>>>(x, pWom, b_off, b_mod, y_om);
  k_deform<<<NTILES, 256, 0, stream>>>(x, y_om, (const unsigned*)pWdef, b_def, x_def);
  k_fuse<<<NTILES, 256, 0, stream>>>(y_std, x_def, (const unsigned*)pWfus, b_fus,
                                     (float*)d_out);
}
